// TS_Encoder_46703474376981
// MI455X (gfx1250) — matrix operand order repaired and checked by reading, not compiled, not run on hardware
//
#include <hip/hip_runtime.h>
#include <cstddef>

// Problem dims
#define B_DIM 512
#define L_DIM 512
#define K_DIM 46
#define H_DIM 128
#define G_DIM 384   // 3*H
#define THREADS 256 // 8 wave32s

// ---------- CDNA5 WMMA types ----------
typedef __attribute__((ext_vector_type(16))) __bf16  v16bf;
typedef __attribute__((ext_vector_type(8)))  float   v8f;
typedef __attribute__((ext_vector_type(4)))  unsigned int uv4;

union FragU {
  v16bf v;
  uv4   q[2];
  unsigned short s[16];
};

__device__ __forceinline__ unsigned short f32_to_bf16(float f) {
  unsigned int u = __float_as_uint(f);
  unsigned int r = (u + 0x7FFFu + ((u >> 16) & 1u)) >> 16; // RNE
  return (unsigned short)r;
}

// fast sigmoid / tanh: v_exp_f32 + v_rcp_f32 only (no IEEE div fixup chain)
__device__ __forceinline__ float sigm(float x) {
  return __builtin_amdgcn_rcpf(1.0f + __expf(-x));
}
__device__ __forceinline__ float fast_tanh(float x) {
  return 1.0f - 2.0f * __builtin_amdgcn_rcpf(1.0f + __expf(2.0f * x));
}

__device__ __forceinline__ v8f wmma_bf16(v16bf a, v16bf b, v8f c) {
  // (neg_a, A, neg_b, B, c_mod, C, reuse_a, reuse_b)
  return __builtin_amdgcn_wmma_f32_16x16x32_bf16(false, a, false, b, (short)0, c, false, false);
}

// A fragment (16x32 bf16, M x K), source row-major [16][stride] in LDS.
// lanes 0-15: K=0..7 (v0..3), K=16..23 (v4..7) of row m=lane; lanes 16-31: K=8..15, K=24..31.
__device__ __forceinline__ v16bf load_frag_a(const unsigned short* base, int stride,
                                             int kBase, int lane) {
  FragU f;
  const int m = lane & 15, half = lane >> 4;
  const unsigned short* row = base + m * stride + kBase + half * 8;
  f.q[0] = *(const uv4*)(row);
  f.q[1] = *(const uv4*)(row + 16);
  return f.v;
}

// B fragment (32x16 bf16, K x N) with B[k][n] = W[n][k], W row-major [n][stride] in global.
// lanes 0-15 hold K=0..7,16..23 of col n=lane, lanes 16-31 hold K=8..15,24..31 -> two b128 loads.
__device__ __forceinline__ v16bf load_frag_b_g(const unsigned short* W, int stride,
                                               int nBase, int kBase, int lane) {
  FragU f;
  const int n = lane & 15, half = lane >> 4;
  const unsigned short* row = W + (size_t)(nBase + n) * stride + kBase + half * 8;
  f.q[0] = *(const uv4*)(row);
  f.q[1] = *(const uv4*)(row + 16);
  return f.v;
}

// C/D f32 16x16: vgpr v -> (m = v + 8*half, n = lane&15)
__device__ __forceinline__ void store_acc(float* acc, int stride, int colBase, int lane, v8f c) {
  const int n = lane & 15, half = lane >> 4;
  #pragma unroll
  for (int v = 0; v < 8; ++v)
    acc[(v + half * 8) * stride + colBase + n] = c[v];
}

// ---------- prep: fp32 weights -> zero-padded bf16 tiles in workspace ----------
__global__ void cvt_pad_kernel(const float* __restrict__ src, unsigned short* __restrict__ dst,
                               int R, int C, int Rp, int Cp) {
  int i = blockIdx.x * blockDim.x + threadIdx.x;
  if (i >= Rp * Cp) return;
  int r = i / Cp, c = i % Cp;
  float v = (r < R && c < C) ? src[r * C + c] : 0.0f;
  dst[i] = f32_to_bf16(v);
}

// Workspace layout (ushort element offsets)
#define OFF_WIH_X 0        // [384][64]
#define OFF_WIH_A 24576
#define OFF_WIH_S 49152
#define OFF_WHH_X 73728    // [384][128]
#define OFF_WHH_A 122880
#define OFF_WHH_S 172032
#define OFF_HRX   221184   // [48][128]
#define OFF_HRA   227328
#define OFF_HR    233472   // [48][96]
#define OFF_SE    238080   // [48][64]

// ---------- recurrent kernel: 32 WGs x 16 batch rows, 8 waves each ----------
__global__ __launch_bounds__(THREADS, 1)
void ts_encoder_kernel(const float* __restrict__ ts, const float* __restrict__ mask,
                       const float* __restrict__ tst,
                       const unsigned short* __restrict__ ws,
                       const float* __restrict__ bihX, const float* __restrict__ bhhX,
                       const float* __restrict__ bihA, const float* __restrict__ bhhA,
                       const float* __restrict__ bihS, const float* __restrict__ bhhS,
                       const float* __restrict__ hrxb, const float* __restrict__ hrab,
                       const float* __restrict__ hrb,  const float* __restrict__ seb,
                       float* __restrict__ out)
{
  // fp32 hidden states (persist across scan)
  __shared__ float s_hx[16][H_DIM], s_ha[16][H_DIM], s_h[16][H_DIM];
  // bf16 WMMA A-operands
  __shared__ unsigned short aHX[16][H_DIM], aHA[16][H_DIM], aHS[16][H_DIM];
  __shared__ unsigned short aXS[16][64], aAL[16][64], aC[16][64], aXADJ[16][64];
  __shared__ unsigned short aCAT[16][96];
  __shared__ float s_ts[16][48];
  __shared__ float s_dec[16];
  // fp32 gate accumulators (gi and gh kept separate for the tanh n-gate)
  __shared__ float giX[16][G_DIM], ghX[16][G_DIM], giA[16][G_DIM], ghA[16][G_DIM];
  __shared__ float accAl[16][48], accS1a[16][48], accS1b[16][48], accXadj[16][48];
  // biases staged in LDS
  __shared__ float sb_ihX[G_DIM], sb_hhX[G_DIM], sb_ihA[G_DIM], sb_hhA[G_DIM];
  __shared__ float sb_ihS[G_DIM], sb_hhS[G_DIM];
  __shared__ float sb_hrx[48], sb_hra[48], sb_hr[48], sb_se[48];

  const int tid  = threadIdx.x;
  const int lane = tid & 31;
  const int wave = tid >> 5;
  const int row0 = blockIdx.x * 16;

  const unsigned short* WIH_X = ws + OFF_WIH_X;
  const unsigned short* WIH_A = ws + OFF_WIH_A;
  const unsigned short* WIH_S = ws + OFF_WIH_S;
  const unsigned short* WHH_X = ws + OFF_WHH_X;
  const unsigned short* WHH_A = ws + OFF_WHH_A;
  const unsigned short* WHH_S = ws + OFF_WHH_S;
  const unsigned short* HRXW  = ws + OFF_HRX;
  const unsigned short* HRAW  = ws + OFF_HRA;
  const unsigned short* HRW   = ws + OFF_HR;
  const unsigned short* SEW   = ws + OFF_SE;

  // init hidden state + zero pad regions + stage biases (once)
  for (int i = tid; i < 16 * H_DIM; i += THREADS) {
    int m = i >> 7, k = i & 127;
    s_hx[m][k] = 0.f; s_ha[m][k] = 0.f; s_h[m][k] = 0.f;
    aHX[m][k] = 0; aHA[m][k] = 0; aHS[m][k] = 0;
  }
  for (int i = tid; i < 16 * 64; i += THREADS) {
    int m = i >> 6, k = i & 63;
    aXS[m][k] = 0; aAL[m][k] = 0; aC[m][k] = 0; aXADJ[m][k] = 0;
  }
  for (int i = tid; i < 16 * 96; i += THREADS) aCAT[i / 96][i % 96] = 0;
  for (int i = tid; i < G_DIM; i += THREADS) {
    sb_ihX[i] = bihX[i]; sb_hhX[i] = bhhX[i];
    sb_ihA[i] = bihA[i]; sb_hhA[i] = bhhA[i];
    sb_ihS[i] = bihS[i]; sb_hhS[i] = bhhS[i];
  }
  if (tid < 48) {
    int k = tid;
    sb_hrx[k] = (k < K_DIM) ? hrxb[k] : 0.f;
    sb_hra[k] = (k < K_DIM) ? hrab[k] : 0.f;
    sb_hr[k]  = (k < K_DIM) ? hrb[k]  : 0.f;
    sb_se[k]  = (k < K_DIM) ? seb[k]  : 0.f;
  }
  __syncthreads();

  for (int t = 0; t < L_DIM; ++t) {
    // ---- Phase A: load ts/mask row, c = 1-mask, x_s = ts*relu(c), decay ----
    for (int i = tid; i < 16 * 64; i += THREADS) {
      int m = i >> 6, k = i & 63;
      if (k < K_DIM) {
        size_t idx = ((size_t)(row0 + m) * L_DIM + t) * K_DIM + k;
        float tsv = ts[idx];
        float c   = 1.f - mask[idx];
        s_ts[m][k] = tsv;
        aC[m][k]   = f32_to_bf16(c);
        aXS[m][k]  = f32_to_bf16(tsv * (c > 0.f ? c : 0.f));
        if ((k & 15) == 0 && t + 1 < L_DIM) {     // pull next step into cache
          __builtin_prefetch(&ts[idx + K_DIM], 0, 3);
          __builtin_prefetch(&mask[idx + K_DIM], 0, 3);
        }
      }
    }
    if (tid < 16) {
      float tc = tst[(size_t)(row0 + tid) * L_DIM + t];
      float dt = (t > 0) ? fabsf(tc - tst[(size_t)(row0 + tid) * L_DIM + t - 1]) : 0.f;
      s_dec[tid] = __builtin_amdgcn_rcpf(__logf(2.718281828459045f + dt));
    }
    __syncthreads();

    // ---- Phase B: alpha pre-activation = c @ se_w^T  (3 N-tiles) ----
    if (wave < 3) {
      v8f acc = {};
      #pragma unroll
      for (int ksb = 0; ksb < 2; ++ksb)
        acc = wmma_bf16(load_frag_a(&aC[0][0], 64, ksb * 32, lane),
                        load_frag_b_g(SEW, 64, wave * 16, ksb * 32, lane), acc);
      store_acc(&accAl[0][0], 48, wave * 16, lane, acc);
    }
    __syncthreads();

    // ---- Phase C: alpha, x_u into concat buffer; decayed hidden bf16 ----
    for (int i = tid; i < 16 * 64; i += THREADS) {
      int m = i >> 6, k = i & 63;
      if (k < K_DIM) {
        float al = sigm(accAl[m][k] + sb_se[k]);
        aAL[m][k] = f32_to_bf16(al);
        aCAT[m][K_DIM + k] = f32_to_bf16(s_ts[m][k] * al);  // x_u
      }
    }
    for (int i = tid; i < 16 * H_DIM; i += THREADS) {
      int m = i >> 7, k = i & 127;
      float d = s_dec[m];
      aHX[m][k] = f32_to_bf16(s_hx[m][k] * d);
      aHA[m][k] = f32_to_bf16(s_ha[m][k] * d);
      aHS[m][k] = f32_to_bf16(s_h[m][k]);   // gru_s hidden is NOT decayed
    }
    __syncthreads();

    // ---- Phase D: gru_x (waves 0-3) & gru_a (waves 4-7) gates ----
    // A fragments hoisted to registers, reused across 6 N-tiles per wave.
    {
      const int g  = wave >> 2;
      const int w4 = wave & 3;
      const unsigned short* Wih = g ? WIH_A : WIH_X;
      const unsigned short* Whh = g ? WHH_A : WHH_X;
      const unsigned short* Am  = g ? &aAL[0][0] : &aXS[0][0];
      const unsigned short* Hm  = g ? &aHA[0][0] : &aHX[0][0];
      float* GI = g ? &giA[0][0] : &giX[0][0];
      float* GH = g ? &ghA[0][0] : &ghX[0][0];
      v16bf agi0 = load_frag_a(Am, 64, 0,  lane);
      v16bf agi1 = load_frag_a(Am, 64, 32, lane);
      v16bf agh0 = load_frag_a(Hm, 128, 0,  lane);
      v16bf agh1 = load_frag_a(Hm, 128, 32, lane);
      v16bf agh2 = load_frag_a(Hm, 128, 64, lane);
      v16bf agh3 = load_frag_a(Hm, 128, 96, lane);
      #pragma unroll
      for (int j = 0; j < 6; ++j) {
        const int nt = w4 + 4 * j;
        v8f gi = {};
        gi = wmma_bf16(agi0, load_frag_b_g(Wih, 64, nt * 16, 0,  lane), gi);
        gi = wmma_bf16(agi1, load_frag_b_g(Wih, 64, nt * 16, 32, lane), gi);
        v8f gh = {};
        gh = wmma_bf16(agh0, load_frag_b_g(Whh, 128, nt * 16, 0,  lane), gh);
        gh = wmma_bf16(agh1, load_frag_b_g(Whh, 128, nt * 16, 32, lane), gh);
        gh = wmma_bf16(agh2, load_frag_b_g(Whh, 128, nt * 16, 64, lane), gh);
        gh = wmma_bf16(agh3, load_frag_b_g(Whh, 128, nt * 16, 96, lane), gh);
        store_acc(GI, G_DIM, nt * 16, lane, gi);
        store_acc(GH, G_DIM, nt * 16, lane, gh);
      }
    }
    __syncthreads();

    // ---- Phase E: GRU nonlinearity for h_x, h_a ----
    for (int i = tid; i < 16 * H_DIM; i += THREADS) {
      int m = i >> 7, k = i & 127;
      float d = s_dec[m];
      {
        float r = sigm(giX[m][k] + sb_ihX[k] + ghX[m][k] + sb_hhX[k]);
        float z = sigm(giX[m][128 + k] + sb_ihX[128 + k] + ghX[m][128 + k] + sb_hhX[128 + k]);
        float n = fast_tanh(giX[m][256 + k] + sb_ihX[256 + k] + r * (ghX[m][256 + k] + sb_hhX[256 + k]));
        float hn = (1.f - z) * n + z * (s_hx[m][k] * d);
        s_hx[m][k] = hn;
        aHX[m][k] = f32_to_bf16(hn);
      }
      {
        float r = sigm(giA[m][k] + sb_ihA[k] + ghA[m][k] + sb_hhA[k]);
        float z = sigm(giA[m][128 + k] + sb_ihA[128 + k] + ghA[m][128 + k] + sb_hhA[128 + k]);
        float n = fast_tanh(giA[m][256 + k] + sb_ihA[256 + k] + r * (ghA[m][256 + k] + sb_hhA[256 + k]));
        float hn = (1.f - z) * n + z * (s_ha[m][k] * d);
        s_ha[m][k] = hn;
        aHA[m][k] = f32_to_bf16(hn);
      }
    }
    __syncthreads();

    // ---- Phase F: hrx / hra heads: h_x@hrx_w^T, h_a@hra_w^T (3+3 N-tiles) ----
    if (wave < 6) {
      int g = wave / 3, nt = wave - g * 3;
      const unsigned short* W  = g ? HRAW : HRXW;
      const unsigned short* Am = g ? &aHA[0][0] : &aHX[0][0];
      v8f acc = {};
      #pragma unroll
      for (int ksb = 0; ksb < 4; ++ksb)
        acc = wmma_bf16(load_frag_a(Am, 128, ksb * 32, lane),
                        load_frag_b_g(W, 128, nt * 16, ksb * 32, lane), acc);
      store_acc(g ? &accS1b[0][0] : &accS1a[0][0], 48, nt * 16, lane, acc);
    }
    __syncthreads();

    // ---- Phase G: x_s_1 = (hrx + b) * sigmoid(hra + b) into concat[0:46] ----
    for (int i = tid; i < 16 * 64; i += THREADS) {
      int m = i >> 6, k = i & 63;
      if (k < K_DIM) {
        float v = (accS1a[m][k] + sb_hrx[k]) * sigm(accS1b[m][k] + sb_hra[k]);
        aCAT[m][k] = f32_to_bf16(v);
      }
    }
    __syncthreads();

    // ---- Phase H: x_adj pre-act = concat @ hr_w^T (K=96 padded) ----
    if (wave < 3) {
      v8f acc = {};
      #pragma unroll
      for (int ksb = 0; ksb < 3; ++ksb)
        acc = wmma_bf16(load_frag_a(&aCAT[0][0], 96, ksb * 32, lane),
                        load_frag_b_g(HRW, 96, wave * 16, ksb * 32, lane), acc);
      store_acc(&accXadj[0][0], 48, wave * 16, lane, acc);
    }
    __syncthreads();

    // ---- Phase I: relu -> x_adj bf16 ----
    for (int i = tid; i < 16 * 64; i += THREADS) {
      int m = i >> 6, k = i & 63;
      if (k < K_DIM) {
        float v = accXadj[m][k] + sb_hr[k];
        aXADJ[m][k] = f32_to_bf16(v > 0.f ? v : 0.f);
      }
    }
    __syncthreads();

    // ---- Phase J: gru_s gates (reuse giX/ghX buffers); hoisted A frags ----
    {
      v16bf agi0 = load_frag_a(&aXADJ[0][0], 64, 0,  lane);
      v16bf agi1 = load_frag_a(&aXADJ[0][0], 64, 32, lane);
      v16bf agh0 = load_frag_a(&aHS[0][0], 128, 0,  lane);
      v16bf agh1 = load_frag_a(&aHS[0][0], 128, 32, lane);
      v16bf agh2 = load_frag_a(&aHS[0][0], 128, 64, lane);
      v16bf agh3 = load_frag_a(&aHS[0][0], 128, 96, lane);
      #pragma unroll
      for (int j = 0; j < 3; ++j) {
        const int nt = wave + 8 * j;
        v8f gi = {};
        gi = wmma_bf16(agi0, load_frag_b_g(WIH_S, 64, nt * 16, 0,  lane), gi);
        gi = wmma_bf16(agi1, load_frag_b_g(WIH_S, 64, nt * 16, 32, lane), gi);
        v8f gh = {};
        gh = wmma_bf16(agh0, load_frag_b_g(WHH_S, 128, nt * 16, 0,  lane), gh);
        gh = wmma_bf16(agh1, load_frag_b_g(WHH_S, 128, nt * 16, 32, lane), gh);
        gh = wmma_bf16(agh2, load_frag_b_g(WHH_S, 128, nt * 16, 64, lane), gh);
        gh = wmma_bf16(agh3, load_frag_b_g(WHH_S, 128, nt * 16, 96, lane), gh);
        store_acc(&giX[0][0], G_DIM, nt * 16, lane, gi);
        store_acc(&ghX[0][0], G_DIM, nt * 16, lane, gh);
      }
    }
    __syncthreads();

    // ---- Phase K: gru_s nonlinearity; hm-mask is identity; write output ----
    for (int i = tid; i < 16 * H_DIM; i += THREADS) {
      int m = i >> 7, k = i & 127;
      float r = sigm(giX[m][k] + sb_ihS[k] + ghX[m][k] + sb_hhS[k]);
      float z = sigm(giX[m][128 + k] + sb_ihS[128 + k] + ghX[m][128 + k] + sb_hhS[128 + k]);
      float n = fast_tanh(giX[m][256 + k] + sb_ihS[256 + k] + r * (ghX[m][256 + k] + sb_hhS[256 + k]));
      float hn = (1.f - z) * n + z * s_h[m][k];
      s_h[m][k] = hn;
      out[((size_t)(row0 + m) * L_DIM + t) * H_DIM + k] = hn;
    }
    __syncthreads();
  }
}

// d_in order (setup_inputs dict flattening):
//  0 ts[B,L,K]  1 mask[B,L,K]  2 ts_t[B,L]
//  3 gru_x.w_ih  4 gru_x.w_hh  5 gru_x.b_ih  6 gru_x.b_hh
//  7 gru_a.w_ih  8 gru_a.w_hh  9 gru_a.b_ih 10 gru_a.b_hh
// 11 gru_s.w_ih 12 gru_s.w_hh 13 gru_s.b_ih 14 gru_s.b_hh
// 15 hrx_w 16 hrx_b 17 hra_w 18 hra_b 19 hr_w 20 hr_b 21 se_w 22 se_b
extern "C" void kernel_launch(void* const* d_in, const int* in_sizes, int n_in,
                              void* d_out, int out_size, void* d_ws, size_t ws_size,
                              hipStream_t stream) {
  (void)in_sizes; (void)n_in; (void)out_size; (void)ws_size;
  unsigned short* ws = (unsigned short*)d_ws;

  auto cvt = [&](int idx, int off, int R, int C, int Rp, int Cp) {
    int total = Rp * Cp;
    cvt_pad_kernel<<<(total + 255) / 256, 256, 0, stream>>>(
        (const float*)d_in[idx], ws + off, R, C, Rp, Cp);
  };
  cvt(3,  OFF_WIH_X, 384, 46,  384, 64);
  cvt(7,  OFF_WIH_A, 384, 46,  384, 64);
  cvt(11, OFF_WIH_S, 384, 46,  384, 64);
  cvt(4,  OFF_WHH_X, 384, 128, 384, 128);
  cvt(8,  OFF_WHH_A, 384, 128, 384, 128);
  cvt(12, OFF_WHH_S, 384, 128, 384, 128);
  cvt(15, OFF_HRX,   46, 128,  48, 128);
  cvt(17, OFF_HRA,   46, 128,  48, 128);
  cvt(19, OFF_HR,    46, 92,   48, 96);
  cvt(21, OFF_SE,    46, 46,   48, 64);

  ts_encoder_kernel<<<dim3(B_DIM / 16), dim3(THREADS), 0, stream>>>(
      (const float*)d_in[0], (const float*)d_in[1], (const float*)d_in[2], ws,
      (const float*)d_in[5],  (const float*)d_in[6],
      (const float*)d_in[9],  (const float*)d_in[10],
      (const float*)d_in[13], (const float*)d_in[14],
      (const float*)d_in[16], (const float*)d_in[18],
      (const float*)d_in[20], (const float*)d_in[22],
      (float*)d_out);
}
